// Linear_int8_est_T_50483045597930
// MI455X (gfx1250) — 8-bit matrix operand order repaired and checked by reading, compiled, not run on hardware
//
#include <hip/hip_runtime.h>

typedef __attribute__((ext_vector_type(8))) int v8i;

#define M_DIM 512
#define K_DIM 512
#define N_DIM 512

#define OFF_R   256
#define OFF_XQ  (OFF_R  + 256 * 256 * 2)
#define OFF_WQ  (OFF_XQ + 512 * 512)
#define OFF_XB  (OFF_WQ + 512 * 512)
#define OFF_WB  (OFF_XB + 512 * 512)

__global__ void init_slots_kernel(unsigned* slots) {
    if (threadIdx.x == 0 && blockIdx.x == 0) { slots[0] = 0u; slots[1] = 0u; }
}

__global__ void absmax_kernel(const float* __restrict__ in, int n, unsigned* slot) {
    __shared__ float smax[256];
    float v = 0.0f;
    for (int i = blockIdx.x * blockDim.x + threadIdx.x; i < n; i += gridDim.x * blockDim.x)
        v = fmaxf(v, fabsf(in[i]));
    smax[threadIdx.x] = v;
    __syncthreads();
    for (int s = 128; s > 0; s >>= 1) {
        if ((int)threadIdx.x < s)
            smax[threadIdx.x] = fmaxf(smax[threadIdx.x], smax[threadIdx.x + s]);
        __syncthreads();
    }
    if (threadIdx.x == 0)
        atomicMax(slot, __float_as_uint(smax[0]));
}

__global__ void scale_kernel(const unsigned* __restrict__ maxbits, float* __restrict__ scales) {
    if (threadIdx.x == 0 && blockIdx.x == 0) {
        float mx = __uint_as_float(maxbits[0]);
        float mw = __uint_as_float(maxbits[1]);
        float Tf = 0.95f * 3.0f + 0.05f * mx;
        float Tw = 0.95f * 0.5f + 0.05f * mw;
        scales[0] = Tf / 127.0f;
        scales[1] = Tw / 127.0f;
    }
}

__global__ void quant_kernel(const float* __restrict__ in, const float* __restrict__ scale,
                             signed char* __restrict__ qs, unsigned char* __restrict__ qb, int n) {
    int i = blockIdx.x * blockDim.x + threadIdx.x;
    if (i < n) {
        float q = rintf(in[i] / scale[0]);
        q = fminf(127.0f, fmaxf(-128.0f, q));
        int qi = (int)q;
        qs[i] = (signed char)qi;
        qb[i] = (unsigned char)(qi + 128);
    }
}

__global__ void residual_kernel(const float* __restrict__ lut, _Float16* __restrict__ R16) {
    int i = blockIdx.x * blockDim.x + threadIdx.x;
    int ii = (i >> 8) - 128;
    int jj = (i & 255) - 128;
    R16[i] = (_Float16)(lut[i] - (float)(ii * jj));
}

__global__ void __launch_bounds__(256)
lut_gemm_kernel(const signed char* __restrict__ xq,
                const signed char* __restrict__ wq,
                const unsigned char* __restrict__ xb,
                const unsigned char* __restrict__ wb,
                const _Float16* __restrict__ Rg,
                const float* __restrict__ scales,
                const float* __restrict__ bias,
                float* __restrict__ out) {
    extern __shared__ __align__(16) unsigned char smem_raw[];

    {
        const uint4* src = (const uint4*)Rg;
        uint4* dst = (uint4*)smem_raw;
        for (int i = threadIdx.x; i < 8192; i += 256) dst[i] = src[i];
    }
    __syncthreads();

    const int tid  = threadIdx.x;
    const int wave = tid >> 5;
    const int lane = tid & 31;
    const int nloc = lane & 15;
    const int mh   = lane >> 4;

    const int m0 = blockIdx.x * 128 + wave * 16;
    const int n0 = blockIdx.y * 16;

    const unsigned char* xrow8 = xb + (m0 + mh * 8) * K_DIM;
    const unsigned char* wrowb = wb + (n0 + nloc) * K_DIM;

    v8i c = {0, 0, 0, 0, 0, 0, 0, 0};
    float res[8];
#pragma unroll
    for (int r = 0; r < 8; ++r) res[r] = 0.0f;

#pragma unroll 1
    for (int kc = 0; kc < 8; ++kc) {
        const int k0 = kc * 64;

        const signed char* abase = xq + (m0 + nloc) * K_DIM + k0 + mh * 8;
        uint2 a01 = *(const uint2*)(abase + 0);
        uint2 a23 = *(const uint2*)(abase + 16);
        uint2 a45 = *(const uint2*)(abase + 32);
        uint2 a67 = *(const uint2*)(abase + 48);
        v8i a;
        a[0] = (int)a01.x; a[1] = (int)a01.y;
        a[2] = (int)a23.x; a[3] = (int)a23.y;
        a[4] = (int)a45.x; a[5] = (int)a45.y;
        a[6] = (int)a67.x; a[7] = (int)a67.y;

        const signed char* bbase = wq + (n0 + nloc) * K_DIM + k0 + mh * 8;
        uint2 b01 = *(const uint2*)(bbase + 0);
        uint2 b23 = *(const uint2*)(bbase + 16);
        uint2 b45 = *(const uint2*)(bbase + 32);
        uint2 b67 = *(const uint2*)(bbase + 48);
        v8i b;
        b[0] = (int)b01.x; b[1] = (int)b01.y; b[2] = (int)b23.x; b[3] = (int)b23.y;
        b[4] = (int)b45.x; b[5] = (int)b45.y; b[6] = (int)b67.x; b[7] = (int)b67.y;

        c = __builtin_amdgcn_wmma_i32_16x16x64_iu8(true, a, true, b, c, false, false);

        const unsigned char* xchunk = xrow8 + k0;
        const unsigned char* wchunk = wrowb + k0;
#pragma unroll
        for (int kg = 0; kg < 16; ++kg) {
            unsigned wdw = *(const unsigned*)(wchunk + kg * 4);
            unsigned w0 = (wdw & 0xFFu) << 1;
            unsigned w1 = ((wdw >> 8) & 0xFFu) << 1;
            unsigned w2 = ((wdw >> 16) & 0xFFu) << 1;
            unsigned w3 = (wdw >> 24) << 1;
#pragma unroll
            for (int r = 0; r < 8; ++r) {
                unsigned xdw = *(const unsigned*)(xchunk + r * K_DIM + kg * 4);
                unsigned o0 = ((xdw & 0xFFu) << 9) + w0;
                unsigned o1 = (((xdw >> 8) & 0xFFu) << 9) + w1;
                unsigned o2 = (((xdw >> 16) & 0xFFu) << 9) + w2;
                unsigned o3 = ((xdw >> 24) << 9) + w3;
                float g0 = (float)*(const _Float16*)(smem_raw + o0);
                float g1 = (float)*(const _Float16*)(smem_raw + o1);
                float g2 = (float)*(const _Float16*)(smem_raw + o2);
                float g3 = (float)*(const _Float16*)(smem_raw + o3);
                res[r] += (g0 + g1) + (g2 + g3);
            }
        }
    }

    const float s  = scales[0] * scales[1];
    const float bn = bias[n0 + nloc];
#pragma unroll
    for (int r = 0; r < 8; ++r) {
        int m = m0 + mh * 8 + r;
        out[m * N_DIM + n0 + nloc] = ((float)c[r] + res[r]) * s + bn;
    }
}

extern "C" void kernel_launch(void* const* d_in, const int* in_sizes, int n_in,
                              void* d_out, int out_size, void* d_ws, size_t ws_size,
                              hipStream_t stream) {
    const float* x    = (const float*)d_in[0];
    const float* w    = (const float*)d_in[1];
    const float* bias = (const float*)d_in[2];
    const float* lut  = (const float*)d_in[3];

    unsigned char* ws = (unsigned char*)d_ws;
    unsigned*       maxslots = (unsigned*)ws;
    float*          scales   = (float*)(ws + 8);
    _Float16*       R16      = (_Float16*)(ws + OFF_R);
    signed char*    xq       = (signed char*)(ws + OFF_XQ);
    signed char*    wq       = (signed char*)(ws + OFF_WQ);
    unsigned char*  xbv      = (unsigned char*)(ws + OFF_XB);
    unsigned char*  wbv      = (unsigned char*)(ws + OFF_WB);
    float*          out      = (float*)d_out;

    init_slots_kernel<<<1, 1, 0, stream>>>(maxslots);
    absmax_kernel<<<256, 256, 0, stream>>>(x, M_DIM * K_DIM, &maxslots[0]);
    absmax_kernel<<<256, 256, 0, stream>>>(w, N_DIM * K_DIM, &maxslots[1]);
    scale_kernel<<<1, 1, 0, stream>>>(maxslots, scales);
    quant_kernel<<<(M_DIM * K_DIM + 255) / 256, 256, 0, stream>>>(x, scales + 0, xq, xbv, M_DIM * K_DIM);
    quant_kernel<<<(N_DIM * K_DIM + 255) / 256, 256, 0, stream>>>(w, scales + 1, wq, wbv, N_DIM * K_DIM);
    residual_kernel<<<(256 * 256) / 256, 256, 0, stream>>>(lut, R16);

    dim3 grid(M_DIM / 128, N_DIM / 16);
    lut_gemm_kernel<<<grid, 256, 256 * 256 * sizeof(_Float16), stream>>>(
        xq, wq, xbv, wbv, R16, scales, bias, out);
}
